// TextRNN_32109175505647
// MI455X (gfx1250) — matrix operand order repaired and checked by reading, not compiled, not run on hardware
//
#include <hip/hip_runtime.h>
#include <hip/hip_bf16.h>
#include <math.h>

typedef __attribute__((ext_vector_type(16))) _Float16 v16h;
typedef __attribute__((ext_vector_type(8)))  _Float16 v8h;
typedef __attribute__((ext_vector_type(8)))  float    v8f;

namespace {
constexpr int Bc = 128, Lc = 48, Vc = 32000, Ec = 512, Hc = 200;
constexpr int Xc = Ec + Hc;     // 712 concat width
constexpr int KX = 736;         // 712 padded to mult of 32
constexpr int G3 = 3 * Hc;      // 600 gates
constexpr int N1 = 640;         // 600 padded to mult of 64
constexpr int KR = 224;         // 200 padded to mult of 32
constexpr int Mrows = Lc * Bc;  // 6144 (row m = t*128 + b)
constexpr float BN_EPS = 1e-5f;
}

// ---------------- WMMA fragment loaders (ISA 7.12.2 layouts) ----------------
// A: 16x32 f16 tile from row-major [.., Kpad]. lanes 0-15: K {0..7,16..23},
// lanes 16-31: K {8..15,24..31}; M = lane&15.
__device__ __forceinline__ v16h load_a_frag(const _Float16* __restrict__ A,
                                            int Kpad, int m0, int k0, int lane) {
  int hs = lane >> 4;
  const _Float16* p = A + (size_t)(m0 + (lane & 15)) * Kpad + k0 + hs * 8;
  v8h lo = *(const v8h*)p;          // 16B -> global_load_b128
  v8h hi = *(const v8h*)(p + 16);
  v16h r;
#pragma unroll
  for (int i = 0; i < 8; ++i) { r[i] = lo[i]; r[i + 8] = hi[i]; }
  return r;
}
// B: 32x16 f16 tile, B[k,n] = W[n*Kpad + k] (W row-major [N,Kpad]).
// lanes 0-15: K {0..7,16..23}; lanes 16-31: K {8..15,24..31}; N = lane&15.
__device__ __forceinline__ v16h load_b_frag(const _Float16* __restrict__ W,
                                            int Kpad, int n0, int k0, int lane) {
  int hs = lane >> 4;
  const _Float16* p = W + (size_t)(n0 + (lane & 15)) * Kpad + k0 + hs * 8;
  v8h lo = *(const v8h*)p;
  v8h hi = *(const v8h*)(p + 16);
  v16h r;
#pragma unroll
  for (int i = 0; i < 8; ++i) { r[i] = lo[i]; r[i + 8] = hi[i]; }
  return r;
}

// ---------------- prep: build f16 operands (pad with zeros) ----------------
__global__ void __launch_bounds__(256) prep_x_kernel(const float* __restrict__ y_embed,
                                                     const float* __restrict__ z,
                                                     _Float16* __restrict__ X) {
  int idx = blockIdx.x * blockDim.x + threadIdx.x;
  if (idx >= Mrows * KX) return;
  int m = idx / KX, c = idx % KX;
  int t = m >> 7, b = m & 127;
  float v = 0.f;
  if (c < Ec) { if (t > 0) v = y_embed[((size_t)b * Lc + (t - 1)) * Ec + c]; }
  else if (c < Xc) v = z[b * Hc + (c - Ec)];
  X[idx] = (_Float16)v;
}
__global__ void __launch_bounds__(256) prep_wih_kernel(const float* __restrict__ w_ih,
                                                       _Float16* __restrict__ W) {
  int idx = blockIdx.x * blockDim.x + threadIdx.x;
  if (idx >= N1 * KX) return;
  int n = idx / KX, k = idx % KX;
  W[idx] = (_Float16)((n < G3 && k < Xc) ? w_ih[(size_t)n * Xc + k] : 0.f);
}
__global__ void __launch_bounds__(256) prep_w3_kernel(const float* __restrict__ w3,
                                                      _Float16* __restrict__ W) {
  int idx = blockIdx.x * blockDim.x + threadIdx.x;
  if (idx >= Vc * KR) return;
  int n = idx / KR, k = idx % KR;
  W[idx] = (_Float16)((k < Hc) ? w3[(size_t)n * Hc + k] : 0.f);
}

// ---------------- GEMM 1: GI = X @ w_ih^T + b_ih  (M=6144, N=600, K=736) ----
// 32(M) x 64(N) tile per wave: 2 A-frags x 4 B-frags -> 8 WMMA per k-step.
__global__ void __launch_bounds__(256) gemm_gi_kernel(const _Float16* __restrict__ A,
                                                      const _Float16* __restrict__ W,
                                                      const float* __restrict__ bias,
                                                      float* __restrict__ C) {
  int gid = blockIdx.x * blockDim.x + threadIdx.x;
  int wave = gid >> 5, lane = gid & 31;
  constexpr int MT = Mrows / 32;            // 192
  int m0 = (wave % MT) * 32;
  int n0 = (wave / MT) * 64;
  v8f acc[2][4] = {};
  for (int k0 = 0; k0 < KX; k0 += 32) {
    v16h a0 = load_a_frag(A, KX, m0,      k0, lane);
    v16h a1 = load_a_frag(A, KX, m0 + 16, k0, lane);
#pragma unroll
    for (int j = 0; j < 4; ++j) {
      v16h b = load_b_frag(W, KX, n0 + 16 * j, k0, lane);
      acc[0][j] = __builtin_amdgcn_wmma_f32_16x16x32_f16(false, a0, false, b,
                                                         (short)0, acc[0][j], false, false);
      acc[1][j] = __builtin_amdgcn_wmma_f32_16x16x32_f16(false, a1, false, b,
                                                         (short)0, acc[1][j], false, false);
    }
  }
  int hs = lane >> 4, nl = lane & 15;
#pragma unroll
  for (int h = 0; h < 2; ++h) {
#pragma unroll
    for (int j = 0; j < 4; ++j) {
      int n = n0 + 16 * j + nl;
      if (n < G3) {
        float bv = bias[n];
#pragma unroll
        for (int i = 0; i < 8; ++i) {
          int m = m0 + 16 * h + i + 8 * hs;
          C[(size_t)m * G3 + n] = acc[h][j][i] + bv;
        }
      }
    }
  }
}

// ---------------- GRU scan: ONE kernel, block b owns batch row b -----------
// h[b] recurrence is independent across the batch (BN coupling is in the
// head, computed later), so the whole t-loop runs locally with h in LDS.
__global__ void __launch_bounds__(256) gru_scan_kernel(const float* __restrict__ GI,
                                                       const float* __restrict__ w_hh,
                                                       const float* __restrict__ b_hh,
                                                       float* __restrict__ Hall) {
  __shared__ float hprev[Hc];
  __shared__ float gh[G3];
  int b = blockIdx.x;
  for (int j = threadIdx.x; j < Hc; j += blockDim.x) hprev[j] = 0.f;
  __syncthreads();
  for (int t = 0; t < Lc; ++t) {
    for (int j = threadIdx.x; j < G3; j += blockDim.x) {
      const float* w = w_hh + (size_t)j * Hc;
      float s = b_hh[j];
#pragma unroll 4
      for (int k = 0; k < Hc; ++k) s += hprev[k] * w[k];
      gh[j] = s;
    }
    __syncthreads();
    const float* gi = GI + ((size_t)t * Bc + b) * G3;
    float* hout = Hall + ((size_t)t * Bc + b) * Hc;
    for (int j = threadIdx.x; j < Hc; j += blockDim.x) {
      float r = 1.f / (1.f + expf(-(gi[j] + gh[j])));
      float u = 1.f / (1.f + expf(-(gi[Hc + j] + gh[Hc + j])));
      float n = tanhf(gi[2 * Hc + j] + r * gh[2 * Hc + j]);
      float h = (1.f - u) * n + u * hprev[j];
      h = fminf(2.f, fmaxf(-2.f, h));
      hout[j] = h;
      hprev[j] = h;
    }
    __syncthreads();
  }
}

// ---------------- head: lin, BN stats, tanh, residual ----------------------
__global__ void __launch_bounds__(256) lin_kernel(const float* __restrict__ Hall,
                                                  const float* __restrict__ w1,
                                                  const float* __restrict__ b1,
                                                  float* __restrict__ LIN) {
  int idx = blockIdx.x * blockDim.x + threadIdx.x;
  if (idx >= Mrows * Hc) return;
  int m = idx / Hc, j = idx % Hc;
  const float* h = Hall + (size_t)m * Hc;
  const float* w = w1 + (size_t)j * Hc;
  float s = b1[j];
#pragma unroll 4
  for (int k = 0; k < Hc; ++k) s += h[k] * w[k];
  LIN[idx] = s;
}
__global__ void __launch_bounds__(256) bnstat_kernel(const float* __restrict__ LIN,
                                                     float* __restrict__ MU,
                                                     float* __restrict__ RS) {
  int idx = blockIdx.x * blockDim.x + threadIdx.x;
  if (idx >= Lc * Hc) return;
  int t = idx / Hc, j = idx % Hc;
  const float* base = LIN + (size_t)t * Bc * Hc + j;
  float s = 0.f, sq = 0.f;
  for (int b = 0; b < Bc; ++b) { float v = base[(size_t)b * Hc]; s += v; sq += v * v; }
  float mu = s * (1.f / Bc);
  float var = sq * (1.f / Bc) - mu * mu;
  MU[idx] = mu;
  RS[idx] = rsqrtf(var + BN_EPS);
}
__global__ void __launch_bounds__(256) act_kernel(const float* __restrict__ LIN,
                                                  const float* __restrict__ MU,
                                                  const float* __restrict__ RS,
                                                  const float* __restrict__ gamma,
                                                  const float* __restrict__ beta,
                                                  float* __restrict__ AACT) {
  int idx = blockIdx.x * blockDim.x + threadIdx.x;
  if (idx >= Mrows * Hc) return;
  int m = idx / Hc, j = idx % Hc;
  int t = m >> 7;
  float v = (LIN[idx] - MU[t * Hc + j]) * RS[t * Hc + j] * gamma[j] + beta[j];
  AACT[idx] = tanhf(v);
}
__global__ void __launch_bounds__(256) res_kernel(const float* __restrict__ AACT,
                                                  const float* __restrict__ Hall,
                                                  const float* __restrict__ w2,
                                                  const float* __restrict__ b2,
                                                  _Float16* __restrict__ RES16) {
  int idx = blockIdx.x * blockDim.x + threadIdx.x;
  if (idx >= Mrows * KR) return;
  int m = idx / KR, j = idx % KR;
  if (j >= Hc) { RES16[idx] = (_Float16)0.f; return; }
  const float* a = AACT + (size_t)m * Hc;
  const float* w = w2 + (size_t)j * Hc;
  float s = b2[j] + Hall[(size_t)m * Hc + j];
#pragma unroll 4
  for (int k = 0; k < Hc; ++k) s += a[k] * w[k];
  RES16[idx] = (_Float16)s;
}

// ---------------- GEMM 2: logits = RES @ w3^T + b3 (M=6144,N=32000,K=224) ---
__global__ void __launch_bounds__(256) gemm_logits_kernel(const _Float16* __restrict__ A,
                                                          const _Float16* __restrict__ W,
                                                          const float* __restrict__ b3,
                                                          float* __restrict__ out) {
  int gid = blockIdx.x * blockDim.x + threadIdx.x;
  int wave = gid >> 5, lane = gid & 31;
  constexpr int MT = Mrows / 32;            // 192
  int m0 = (wave % MT) * 32;
  int n0 = (wave / MT) * 64;
  v8f acc[2][4] = {};
  for (int k0 = 0; k0 < KR; k0 += 32) {
    v16h a0 = load_a_frag(A, KR, m0,      k0, lane);
    v16h a1 = load_a_frag(A, KR, m0 + 16, k0, lane);
#pragma unroll
    for (int j = 0; j < 4; ++j) {
      v16h b = load_b_frag(W, KR, n0 + 16 * j, k0, lane);
      acc[0][j] = __builtin_amdgcn_wmma_f32_16x16x32_f16(false, a0, false, b,
                                                         (short)0, acc[0][j], false, false);
      acc[1][j] = __builtin_amdgcn_wmma_f32_16x16x32_f16(false, a1, false, b,
                                                         (short)0, acc[1][j], false, false);
    }
  }
  int hs = lane >> 4, nl = lane & 15;
#pragma unroll
  for (int h = 0; h < 2; ++h) {
#pragma unroll
    for (int j = 0; j < 4; ++j) {
      int n = n0 + 16 * j + nl;
      float bv = b3[n];
#pragma unroll
      for (int i = 0; i < 8; ++i) {
        int m = m0 + 16 * h + i + 8 * hs;
        int t = m >> 7, b = m & 127;
        out[((size_t)b * Lc + t) * Vc + n] = acc[h][j][i] + bv;  // word_preds[b][t][n]
      }
    }
  }
}

// ---------------- single-pass online log-softmax pick ----------------------
__global__ void __launch_bounds__(256) softmax_pick_kernel(const float* __restrict__ out,
                                                           const int* __restrict__ y,
                                                           float* __restrict__ LP) {
  __shared__ float redm[256];
  __shared__ float reds[256];
  int m = blockIdx.x;
  int t = m >> 7, b = m & 127;
  const float* row = out + ((size_t)b * Lc + t) * Vc;
  int tid = threadIdx.x;
  float mx = -3.4e38f, s = 0.f;
  for (int v = tid; v < Vc; v += 256) {          // every thread sees >=125 elems
    float x = row[v];
    float nm = fmaxf(mx, x);
    s = s * expf(mx - nm) + expf(x - nm);
    mx = nm;
  }
  redm[tid] = mx; reds[tid] = s; __syncthreads();
  for (int st = 128; st > 0; st >>= 1) {         // fixed-order merge: deterministic
    if (tid < st) {
      float m1 = redm[tid], s1 = reds[tid];
      float m2 = redm[tid + st], s2 = reds[tid + st];
      float nm = fmaxf(m1, m2);
      reds[tid] = s1 * expf(m1 - nm) + s2 * expf(m2 - nm);
      redm[tid] = nm;
    }
    __syncthreads();
  }
  if (tid == 0) {
    int tgt = y[b * Lc + t];
    LP[m] = row[tgt] - redm[0] - logf(reds[0]);
  }
}
__global__ void final_sum_kernel(const float* __restrict__ LP, float* __restrict__ outlp) {
  int b = threadIdx.x;
  if (b < Bc) {
    float s = 0.f;
    for (int t = 0; t < Lc; ++t) s += LP[t * Bc + b];
    outlp[b] = s;
  }
}

// ---------------- host ------------------------------------------------------
extern "C" void kernel_launch(void* const* d_in, const int* in_sizes, int n_in,
                              void* d_out, int out_size, void* d_ws, size_t ws_size,
                              hipStream_t stream) {
  const float* z       = (const float*)d_in[0];
  const float* y_embed = (const float*)d_in[1];
  const int*   y       = (const int*)d_in[2];
  const float* w_ih    = (const float*)d_in[3];
  const float* w_hh    = (const float*)d_in[4];
  const float* b_ih    = (const float*)d_in[5];
  const float* b_hh    = (const float*)d_in[6];
  const float* w1      = (const float*)d_in[7];
  const float* b1      = (const float*)d_in[8];
  const float* gamma   = (const float*)d_in[9];
  const float* beta    = (const float*)d_in[10];
  const float* w2      = (const float*)d_in[11];
  const float* b2      = (const float*)d_in[12];
  const float* w3      = (const float*)d_in[13];
  const float* b3      = (const float*)d_in[14];
  float* out = (float*)d_out;

  // workspace carve-up (256B aligned)
  char* ws = (char*)d_ws;
  size_t off = 0;
  auto carve = [&](size_t bytes) { char* p = ws + off; off = (off + bytes + 255) & ~(size_t)255; return p; };
  _Float16* Xf16  = (_Float16*)carve((size_t)Mrows * KX * 2);
  _Float16* Wih16 = (_Float16*)carve((size_t)N1 * KX * 2);
  _Float16* W316  = (_Float16*)carve((size_t)Vc * KR * 2);
  float*    GI    = (float*)carve((size_t)Mrows * G3 * 4);
  float*    Hall  = (float*)carve((size_t)Mrows * Hc * 4);
  float*    LIN   = (float*)carve((size_t)Mrows * Hc * 4);
  float*    AACT  = (float*)carve((size_t)Mrows * Hc * 4);
  float*    MU    = (float*)carve((size_t)Lc * Hc * 4);
  float*    RS    = (float*)carve((size_t)Lc * Hc * 4);
  _Float16* RES16 = (_Float16*)carve((size_t)Mrows * KR * 2);
  float*    LP    = (float*)carve((size_t)Mrows * 4);
  (void)ws_size; (void)n_in; (void)in_sizes; (void)out_size;

  // 1) build f16 operands
  prep_x_kernel<<<(Mrows * KX) / 256, 256, 0, stream>>>(y_embed, z, Xf16);
  prep_wih_kernel<<<(N1 * KX) / 256, 256, 0, stream>>>(w_ih, Wih16);
  prep_w3_kernel<<<(Vc * KR) / 256, 256, 0, stream>>>(w3, W316);

  // 2) GI = X @ w_ih^T + b_ih (hoisted out of the scan) — WMMA, 32x64 tiles
  {
    int strips = (Mrows / 32) * (N1 / 64);           // 1920 waves
    gemm_gi_kernel<<<strips * 32 / 256, 256, 0, stream>>>(Xf16, Wih16, b_ih, GI);
  }

  // 3) full GRU scan in one launch (block b = batch row b, t-loop in LDS)
  gru_scan_kernel<<<Bc, 256, 0, stream>>>(GI, w_hh, b_hh, Hall);

  // 4) head (parallel over all (t,b))
  lin_kernel<<<(Mrows * Hc) / 256, 256, 0, stream>>>(Hall, w1, b1, LIN);
  bnstat_kernel<<<(Lc * Hc + 255) / 256, 256, 0, stream>>>(LIN, MU, RS);
  act_kernel<<<(Mrows * Hc) / 256, 256, 0, stream>>>(LIN, MU, RS, gamma, beta, AACT);
  res_kernel<<<(Mrows * KR) / 256, 256, 0, stream>>>(AACT, Hall, w2, b2, RES16);

  // 5) logits = RES @ w3^T + b3 — the big WMMA GEMM, writes word_preds
  {
    int strips = (Mrows / 32) * (Vc / 64);           // 96000 waves
    gemm_logits_kernel<<<strips * 32 / 256, 256, 0, stream>>>(RES16, W316, b3, out);
  }

  // 6) online log-softmax gather + deterministic per-batch sum
  softmax_pick_kernel<<<Mrows, 256, 0, stream>>>(out, y, LP);
  final_sum_kernel<<<1, 128, 0, stream>>>(LP, out + (size_t)Bc * Lc * Vc);
}
